// DySAT_85856396247960
// MI455X (gfx1250) — matrix operand order repaired and checked by reading, not compiled, not run on hardware
//
#include <hip/hip_runtime.h>

#define Nn   50000
#define Ee   500000
#define Tt   8
#define FIN  128
#define Dm   256
#define Hh   32
#define Cc   8
#define OUTC 64
#define EPSF 1e-5f

typedef __attribute__((ext_vector_type(16))) __bf16 v16bf;
typedef __attribute__((ext_vector_type(4)))  __bf16 v4bf;
typedef __attribute__((ext_vector_type(8)))  float  v8f;

union AFrag { uint4 u[2]; v16bf v; };

// ---------------------------------------------------------------------------
// WMMA bf16 GEMM:  Y[M,Nc] = X[M,K] @ W[Nc,K]^T (+ bias)
// X fp32 (native-cvt to bf16 while staging in LDS); W bf16, staged into LDS
// via GLOBAL_LOAD_ASYNC_TO_LDS_B128 (ASYNCcnt DMA) overlapping the A-stage.
// Block: 256 threads = 8 waves; tile BM=128, BN=64, BK=32.
// Wave (wm 0..3, wn 0..1) computes a 32x32 patch = 4 x v_wmma_f32_16x16x32_bf16
// ---------------------------------------------------------------------------
__global__ __launch_bounds__(256, 2)
void gemm_wmma_bf16(const float* __restrict__ X, const unsigned short* __restrict__ W,
                    const float* __restrict__ bias, float* __restrict__ Y,
                    int M, int K, int Nc)
{
    __shared__ __align__(16) unsigned short As[128][32];
    __shared__ __align__(16) unsigned short Bs[64][32];

    const int tid  = threadIdx.x;
    const int lane = tid & 31;
    const int wave = tid >> 5;
    const int wm   = wave >> 1;        // 0..3
    const int wn   = wave & 1;         // 0..1
    const int lm   = lane & 15;
    const int blockRow = blockIdx.x * 128;
    const int blockCol = blockIdx.y * 64;

    v8f zero = {0.f,0.f,0.f,0.f,0.f,0.f,0.f,0.f};
    v8f acc00 = zero, acc01 = zero, acc10 = zero, acc11 = zero;

    const int arow = tid >> 3;         // 0..31 per sub-iter
    const int acol = (tid & 7) * 4;    // 0,4,..,28
    const int brow = tid >> 2;         // 0..63
    const int bcol = (tid & 3) * 8;    // 0,8,16,24

    // LDS byte offset of this thread's B-tile slot (low 32 bits of generic ptr)
    const unsigned ldsb = (unsigned)(unsigned long long)&Bs[brow][bcol];

    for (int k0 = 0; k0 < K; k0 += 32) {
        // ---- B tile: async DMA global->LDS (16B per lane), overlaps A staging
        {
            const unsigned short* wp = W + (size_t)(blockCol + brow) * K + k0 + bcol;
            asm volatile("global_load_async_to_lds_b128 %0, %1, off"
                         :: "v"(ldsb), "v"(wp)
                         : "memory");
        }
        // ---- A tile: fp32 -> bf16 (native cvt) : 128x32
        #pragma unroll
        for (int it = 0; it < 4; ++it) {
            int r  = it * 32 + arow;
            int gr = blockRow + r;
            float4 f = make_float4(0.f, 0.f, 0.f, 0.f);
            if (gr < M) {
                const float* xp = X + (size_t)gr * K + k0 + acol;
                f = *(const float4*)xp;
                if (k0 + 32 < K) __builtin_prefetch(xp + 32, 0, 3);  // global_prefetch_b8
            }
            v4bf pk;
            pk.x = (__bf16)f.x; pk.y = (__bf16)f.y;
            pk.z = (__bf16)f.z; pk.w = (__bf16)f.w;
            *(v4bf*)&As[r][acol] = pk;
        }
        asm volatile("s_wait_asynccnt 0x0" ::: "memory");
        __syncthreads();

        // ---- fragments (ISA 16-bit A 16x32 / B 32x16 layouts)
        const int kbA = (lane < 16) ? 0 : 8;   // K-halves per lane group
        const int kbB = (lane < 16) ? 0 : 8;
        AFrag a0, a1, b0, b1;
        {
            int r0 = wm * 32 + lm;
            int r1 = r0 + 16;
            a0.u[0] = *(const uint4*)&As[r0][kbA];
            a0.u[1] = *(const uint4*)&As[r0][kbA + 16];
            a1.u[0] = *(const uint4*)&As[r1][kbA];
            a1.u[1] = *(const uint4*)&As[r1][kbA + 16];
            int c0 = wn * 32 + lm;
            int c1 = c0 + 16;
            b0.u[0] = *(const uint4*)&Bs[c0][kbB];
            b0.u[1] = *(const uint4*)&Bs[c0][kbB + 16];
            b1.u[0] = *(const uint4*)&Bs[c1][kbB];
            b1.u[1] = *(const uint4*)&Bs[c1][kbB + 16];
        }
        acc00 = __builtin_amdgcn_wmma_f32_16x16x32_bf16(false, a0.v, false, b0.v, (short)0, acc00, false, false);
        acc01 = __builtin_amdgcn_wmma_f32_16x16x32_bf16(false, a0.v, false, b1.v, (short)0, acc01, false, false);
        acc10 = __builtin_amdgcn_wmma_f32_16x16x32_bf16(false, a1.v, false, b0.v, (short)0, acc10, false, false);
        acc11 = __builtin_amdgcn_wmma_f32_16x16x32_bf16(false, a1.v, false, b1.v, (short)0, acc11, false, false);
        __syncthreads();
    }

    // ---- epilogue: C layout: VGPR i -> (lane<16: M=i, N=lane) (lane>=16: M=8+i, N=lane-16)
    const int rbase = blockRow + wm * 32 + ((lane >> 4) ? 8 : 0);
    const int c0 = blockCol + wn * 32 + lm;
    const int c1 = c0 + 16;
    const float bv0 = bias ? bias[c0] : 0.f;
    const float bv1 = bias ? bias[c1] : 0.f;
    #pragma unroll
    for (int i = 0; i < 8; ++i) {
        int r0 = rbase + i;
        int r1 = r0 + 16;
        if (r0 < M) {
            Y[(size_t)r0 * Nc + c0] = acc00[i] + bv0;
            Y[(size_t)r0 * Nc + c1] = acc01[i] + bv1;
        }
        if (r1 < M) {
            Y[(size_t)r1 * Nc + c0] = acc10[i] + bv0;
            Y[(size_t)r1 * Nc + c1] = acc11[i] + bv1;
        }
    }
}

// ---------------------------------------------------------------------------
// Small utility kernels
// ---------------------------------------------------------------------------
__global__ void cvt_bf16_kernel(const float* __restrict__ s, unsigned short* __restrict__ d, long n) {
    long i = (long)blockIdx.x * blockDim.x + threadIdx.x;
    if (i < n) {
        __bf16 b = (__bf16)s[i];
        d[i] = __builtin_bit_cast(unsigned short, b);
    }
}

__global__ void fill_kernel(float* __restrict__ p, float v, long n) {
    long i = (long)blockIdx.x * blockDim.x + threadIdx.x;
    if (i < n) p[i] = v;
}

__global__ void bcast256_kernel(float* __restrict__ o, const float* __restrict__ b, long n) {
    long i = (long)blockIdx.x * blockDim.x + threadIdx.x;
    if (i < n) o[i] = b[i & 255];
}

// es/ed: per-(node,head) dot of h[n,head,:] with a_s/a_d[head,:]
__global__ void attn_coef_kernel(const float* __restrict__ h, const float* __restrict__ a_s,
                                 const float* __restrict__ a_d, float* __restrict__ es,
                                 float* __restrict__ ed, long n_total) {
    long i = (long)blockIdx.x * blockDim.x + threadIdx.x;
    if (i >= n_total) return;
    int hd = (int)(i & 31);
    const float* hp = h + i * 8;     // i = n*32+head -> i*8 = n*256 + head*8
    float s = 0.f, d = 0.f;
    #pragma unroll
    for (int c = 0; c < 8; ++c) {
        float v = hp[c];
        s += v * a_s[hd * 8 + c];
        d += v * a_d[hd * 8 + c];
    }
    es[i] = s; ed[i] = d;
}

__device__ __forceinline__ float edge_logit(const float* es, const float* ed, int s, int d, int h) {
    float v = es[(size_t)s * 32 + h] + ed[(size_t)d * 32 + h];
    return v > 0.f ? v : 0.2f * v;    // LeakyReLU(0.2)
}

__device__ __forceinline__ void atomicMaxF(float* addr, float val) {
    int* ai = (int*)addr;
    int old = __float_as_int(*addr);
    while (__int_as_float(old) < val) {
        int prev = atomicCAS(ai, old, __float_as_int(val));
        if (prev == old) break;
        old = prev;
    }
}

__global__ void edge_max_kernel(const int* __restrict__ src, const int* __restrict__ dst,
                                const float* __restrict__ es, const float* __restrict__ ed,
                                float* __restrict__ m) {
    int e = blockIdx.x * blockDim.x + threadIdx.x;
    if (e >= Ee + Nn) return;
    int s, d;
    if (e < Ee) { s = src[e]; d = dst[e]; } else { s = e - Ee; d = s; }  // self-loops
    for (int h = 0; h < Hh; ++h)
        atomicMaxF(&m[(size_t)d * 32 + h], edge_logit(es, ed, s, d, h));
}

__global__ void edge_sum_kernel(const int* __restrict__ src, const int* __restrict__ dst,
                                const float* __restrict__ es, const float* __restrict__ ed,
                                const float* __restrict__ m, float* __restrict__ den) {
    int e = blockIdx.x * blockDim.x + threadIdx.x;
    if (e >= Ee + Nn) return;
    int s, d;
    if (e < Ee) { s = src[e]; d = dst[e]; } else { s = e - Ee; d = s; }
    for (int h = 0; h < Hh; ++h) {
        float v = edge_logit(es, ed, s, d, h);
        atomicAdd(&den[(size_t)d * 32 + h], __expf(v - m[(size_t)d * 32 + h]));
    }
}

__global__ void edge_scatter_kernel(const int* __restrict__ src, const int* __restrict__ dst,
                                    const float* __restrict__ es, const float* __restrict__ ed,
                                    const float* __restrict__ m, const float* __restrict__ den,
                                    const float* __restrict__ h, float* __restrict__ out) {
    int e = blockIdx.x * blockDim.x + threadIdx.x;
    if (e >= Ee + Nn) return;
    int s, d;
    if (e < Ee) { s = src[e]; d = dst[e]; } else { s = e - Ee; d = s; }
    for (int hd = 0; hd < Hh; ++hd) {
        float v  = edge_logit(es, ed, s, d, hd);
        float ex = __expf(v - m[(size_t)d * 32 + hd]);
        float alpha = ex / den[(size_t)d * 32 + hd];
        const float* hs = h + (size_t)s * Dm + hd * 8;
        float* op = out + (size_t)d * Dm + hd * 8;
        #pragma unroll
        for (int c = 0; c < 8; ++c) atomicAdd(&op[c], alpha * hs[c]);
    }
}

// LayerNorm over D=256 + ReLU, one wave per node (wave32: 8 elems/lane)
__global__ __launch_bounds__(256)
void ln_relu_kernel(const float* __restrict__ x, const float* __restrict__ g,
                    const float* __restrict__ b, float* __restrict__ y, int M) {
    int gid = blockIdx.x * blockDim.x + threadIdx.x;
    int n = gid >> 5, lane = gid & 31;
    if (n >= M) return;
    const float* xr = x + (size_t)n * Dm + lane * 8;
    float v[8]; float s = 0.f;
    #pragma unroll
    for (int i = 0; i < 8; ++i) { v[i] = xr[i]; s += v[i]; }
    #pragma unroll
    for (int o = 16; o; o >>= 1) s += __shfl_xor(s, o);
    float mu = s * (1.f / Dm);
    float q = 0.f;
    #pragma unroll
    for (int i = 0; i < 8; ++i) { float t = v[i] - mu; q += t * t; }
    #pragma unroll
    for (int o = 16; o; o >>= 1) q += __shfl_xor(q, o);
    float inv = rsqrtf(q * (1.f / Dm) + EPSF);
    float* yr = y + (size_t)n * Dm + lane * 8;
    #pragma unroll
    for (int i = 0; i < 8; ++i) {
        float t = (v[i] - mu) * inv * g[lane * 8 + i] + b[lane * 8 + i];
        yr[i] = t > 0.f ? t : 0.f;
    }
}

// scores[n,s] = dot(q_last[n], k_s[n]) / 16 ; one wave per node
__global__ __launch_bounds__(256)
void score_kernel(const float* __restrict__ q, const float* __restrict__ k,
                  float* __restrict__ wts, int s, int M) {
    int gid = blockIdx.x * blockDim.x + threadIdx.x;
    int n = gid >> 5, lane = gid & 31;
    if (n >= M) return;
    const float* qp = q + (size_t)n * Dm + lane * 8;
    const float* kp = k + (size_t)n * Dm + lane * 8;
    float acc = 0.f;
    #pragma unroll
    for (int i = 0; i < 8; ++i) acc += qp[i] * kp[i];
    #pragma unroll
    for (int o = 16; o; o >>= 1) acc += __shfl_xor(acc, o);
    if (lane == 0) wts[(size_t)n * Tt + s] = acc * 0.0625f;   // 1/sqrt(256)
}

__global__ void softmax8_kernel(float* __restrict__ w, int M) {
    int n = blockIdx.x * blockDim.x + threadIdx.x;
    if (n >= M) return;
    float* p = w + (size_t)n * Tt;
    float mx = p[0];
    #pragma unroll
    for (int i = 1; i < Tt; ++i) mx = fmaxf(mx, p[i]);
    float e[Tt]; float s = 0.f;
    #pragma unroll
    for (int i = 0; i < Tt; ++i) { e[i] = __expf(p[i] - mx); s += e[i]; }
    float inv = 1.f / s;
    #pragma unroll
    for (int i = 0; i < Tt; ++i) p[i] = e[i] * inv;
}

__global__ void axpy_acc_kernel(float* __restrict__ attn, const float* __restrict__ v,
                                const float* __restrict__ w, int s, long n) {
    long i = (long)blockIdx.x * blockDim.x + threadIdx.x;
    if (i < n) attn[i] += w[(i >> 8) * Tt + s] * v[i];
}

// BatchNorm column stats over N for 64 cols: acc[0..63]=sum, acc[64..127]=sumsq
__global__ __launch_bounds__(256)
void bn_stats_kernel(const float* __restrict__ lg, float* __restrict__ acc, int M) {
    int c = threadIdx.x & 63;
    int rsub = threadIdx.x >> 6;  // 0..3
    int rbeg = blockIdx.x * 1024 + rsub;
    int rend = (blockIdx.x + 1) * 1024; if (rend > M) rend = M;
    float s = 0.f, q = 0.f;
    for (int r = rbeg; r < rend; r += 4) {
        float v = lg[(size_t)r * OUTC + c];
        s += v; q += v * v;
    }
    __shared__ float sh[256], shq[256];
    sh[threadIdx.x] = s; shq[threadIdx.x] = q;
    __syncthreads();
    if (rsub == 0) {
        s = sh[c] + sh[64 + c] + sh[128 + c] + sh[192 + c];
        q = shq[c] + shq[64 + c] + shq[128 + c] + shq[192 + c];
        atomicAdd(&acc[c], s);
        atomicAdd(&acc[64 + c], q);
    }
}

// BN normalize + log_softmax per row; one wave per row (lane handles 2 cols)
__global__ __launch_bounds__(256)
void bn_final_kernel(const float* __restrict__ lg, const float* __restrict__ acc,
                     const float* __restrict__ g, const float* __restrict__ b,
                     float* __restrict__ out, int M) {
    int gid = blockIdx.x * blockDim.x + threadIdx.x;
    int n = gid >> 5, lane = gid & 31;
    if (n >= M) return;
    float y[2]; float mx = -3.4e38f;
    #pragma unroll
    for (int i = 0; i < 2; ++i) {
        int c = lane + i * 32;
        float mu  = acc[c] * (1.f / M);
        float var = acc[64 + c] * (1.f / M) - mu * mu;
        float v = (lg[(size_t)n * OUTC + c] - mu) * rsqrtf(var + EPSF) * g[c] + b[c];
        y[i] = v; mx = fmaxf(mx, v);
    }
    #pragma unroll
    for (int o = 16; o; o >>= 1) mx = fmaxf(mx, __shfl_xor(mx, o));
    float se = __expf(y[0] - mx) + __expf(y[1] - mx);
    #pragma unroll
    for (int o = 16; o; o >>= 1) se += __shfl_xor(se, o);
    float lse = mx + logf(se);
    out[(size_t)n * OUTC + lane]      = y[0] - lse;
    out[(size_t)n * OUTC + lane + 32] = y[1] - lse;
}

// ---------------------------------------------------------------------------
extern "C" void kernel_launch(void* const* d_in, const int* in_sizes, int n_in,
                              void* d_out, int out_size, void* d_ws, size_t ws_size,
                              hipStream_t stream) {
    const float* feats = (const float*)d_in[0];
    const int*   ei    = (const int*)d_in[1];
    const float* W1    = (const float*)d_in[2];
    const float* a_s1  = (const float*)d_in[3];
    const float* a_d1  = (const float*)d_in[4];
    const float* b1    = (const float*)d_in[5];
    const float* W2    = (const float*)d_in[6];
    const float* a_s2  = (const float*)d_in[7];
    const float* a_d2  = (const float*)d_in[8];
    const float* b2    = (const float*)d_in[9];
    const float* ln_g  = (const float*)d_in[10];
    const float* ln_b  = (const float*)d_in[11];
    const float* Wqkv  = (const float*)d_in[12];
    const float* bqkv  = (const float*)d_in[13];
    const float* Wo    = (const float*)d_in[14];
    const float* bo    = (const float*)d_in[15];
    const float* Wout  = (const float*)d_in[16];
    const float* bout  = (const float*)d_in[17];
    const float* bn_g  = (const float*)d_in[18];
    const float* bn_b  = (const float*)d_in[19];

    // ---- workspace carve-up (256B aligned)
    char* ws = (char*)d_ws;
    size_t off = 0;
    auto alloc = [&](size_t bytes) -> char* {
        char* p = ws + off;
        off = (off + bytes + 255) & ~(size_t)255;
        return p;
    };
    unsigned short* wb1   = (unsigned short*)alloc((size_t)Dm * FIN * 2);
    unsigned short* wb2   = (unsigned short*)alloc((size_t)Dm * Dm * 2);
    unsigned short* wbq   = (unsigned short*)alloc((size_t)Dm * Dm * 2);
    unsigned short* wbk   = (unsigned short*)alloc((size_t)Dm * Dm * 2);
    unsigned short* wbv   = (unsigned short*)alloc((size_t)Dm * Dm * 2);
    unsigned short* wbo   = (unsigned short*)alloc((size_t)Dm * Dm * 2);
    unsigned short* wbout = (unsigned short*)alloc((size_t)OUTC * Dm * 2);
    float* hbuf  = (float*)alloc((size_t)Nn * Dm * 4);
    float* xbuf  = (float*)alloc((size_t)Nn * Dm * 4);
    float* esb   = (float*)alloc((size_t)Nn * Hh * 4);
    float* edb   = (float*)alloc((size_t)Nn * Hh * 4);
    float* mbuf  = (float*)alloc((size_t)Nn * Hh * 4);
    float* dbuf  = (float*)alloc((size_t)Nn * Hh * 4);
    float* embf  = (float*)alloc((size_t)Tt * Nn * Dm * 4);
    float* qlast = (float*)alloc((size_t)Nn * Dm * 4);
    float* tmpb  = (float*)alloc((size_t)Nn * Dm * 4);
    float* attnb = (float*)alloc((size_t)Nn * Dm * 4);
    float* obuf  = (float*)alloc((size_t)Nn * Dm * 4);
    float* lgts  = (float*)alloc((size_t)Nn * OUTC * 4);
    float* wtsb  = (float*)alloc((size_t)Nn * Tt * 4);
    float* bnacc = (float*)alloc(128 * 4);

    auto blk = [](long n) { return (unsigned)((n + 255) / 256); };
    auto cvt = [&](const float* s, unsigned short* d, long n) {
        cvt_bf16_kernel<<<blk(n), 256, 0, stream>>>(s, d, n);
    };
    auto fill = [&](float* p, float v, long n) {
        fill_kernel<<<blk(n), 256, 0, stream>>>(p, v, n);
    };
    auto gemm = [&](const float* X, const unsigned short* W, const float* bias,
                    float* Y, int M, int K, int Nc) {
        dim3 g((M + 127) / 128, Nc / 64);
        gemm_wmma_bf16<<<g, 256, 0, stream>>>(X, W, bias, Y, M, K, Nc);
    };

    // ---- weights -> bf16
    cvt(W1, wb1, (long)Dm * FIN);
    cvt(W2, wb2, (long)Dm * Dm);
    cvt(Wqkv,                    wbq, (long)Dm * Dm);
    cvt(Wqkv + (size_t)Dm * Dm,  wbk, (long)Dm * Dm);
    cvt(Wqkv + (size_t)2*Dm*Dm,  wbv, (long)Dm * Dm);
    cvt(Wo, wbo, (long)Dm * Dm);
    cvt(Wout, wbout, (long)OUTC * Dm);

    const long nh  = (long)Nn * Hh;
    const long nd  = (long)Nn * Dm;
    const unsigned eblk = blk((long)Ee + Nn);
    const unsigned wblk = blk((long)Nn * 32);   // wave-per-node kernels

    // ---- per-time-step GAT stack
    for (int t = 0; t < Tt; ++t) {
        const int* src = ei + (size_t)t * 2 * Ee;
        const int* dst = src + Ee;

        // layer 1 (input: feats[t], K=128)
        gemm(feats + (size_t)t * Nn * FIN, wb1, nullptr, hbuf, Nn, FIN, Dm);
        attn_coef_kernel<<<blk(nh), 256, 0, stream>>>(hbuf, a_s1, a_d1, esb, edb, nh);
        fill(mbuf, -1e30f, nh);
        fill(dbuf, 0.f, nh);
        edge_max_kernel<<<eblk, 256, 0, stream>>>(src, dst, esb, edb, mbuf);
        edge_sum_kernel<<<eblk, 256, 0, stream>>>(src, dst, esb, edb, mbuf, dbuf);
        bcast256_kernel<<<blk(nd), 256, 0, stream>>>(xbuf, b1, nd);
        edge_scatter_kernel<<<eblk, 256, 0, stream>>>(src, dst, esb, edb, mbuf, dbuf, hbuf, xbuf);

        // layer 2 (input: xbuf, K=256)
        gemm(xbuf, wb2, nullptr, hbuf, Nn, Dm, Dm);
        attn_coef_kernel<<<blk(nh), 256, 0, stream>>>(hbuf, a_s2, a_d2, esb, edb, nh);
        fill(mbuf, -1e30f, nh);
        fill(dbuf, 0.f, nh);
        edge_max_kernel<<<eblk, 256, 0, stream>>>(src, dst, esb, edb, mbuf);
        edge_sum_kernel<<<eblk, 256, 0, stream>>>(src, dst, esb, edb, mbuf, dbuf);
        bcast256_kernel<<<blk(nd), 256, 0, stream>>>(xbuf, b2, nd);
        edge_scatter_kernel<<<eblk, 256, 0, stream>>>(src, dst, esb, edb, mbuf, dbuf, hbuf, xbuf);

        // LayerNorm + ReLU -> emb[t]
        ln_relu_kernel<<<wblk, 256, 0, stream>>>(xbuf, ln_g, ln_b,
                                                 embf + (size_t)t * Nn * Dm, Nn);
    }

    // ---- temporal attention (only t = T-1 output is consumed downstream)
    const float* bq = bqkv;
    const float* bk = bqkv + Dm;
    const float* bv = bqkv + 2 * Dm;

    gemm(embf + (size_t)(Tt - 1) * Nn * Dm, wbq, bq, qlast, Nn, Dm, Dm);
    for (int s = 0; s < Tt; ++s) {
        gemm(embf + (size_t)s * Nn * Dm, wbk, bk, tmpb, Nn, Dm, Dm);
        score_kernel<<<wblk, 256, 0, stream>>>(qlast, tmpb, wtsb, s, Nn);
    }
    softmax8_kernel<<<blk(Nn), 256, 0, stream>>>(wtsb, Nn);
    fill(attnb, 0.f, nd);
    for (int s = 0; s < Tt; ++s) {
        gemm(embf + (size_t)s * Nn * Dm, wbv, bv, tmpb, Nn, Dm, Dm);
        axpy_acc_kernel<<<blk(nd), 256, 0, stream>>>(attnb, tmpb, wtsb, s, nd);
    }
    gemm(attnb, wbo, bo, obuf, Nn, Dm, Dm);

    // ---- output head: logits -> BatchNorm -> log_softmax
    gemm(obuf, wbout, bout, lgts, Nn, Dm, OUTC);
    fill(bnacc, 0.f, 128);
    bn_stats_kernel<<<(Nn + 1023) / 1024, 256, 0, stream>>>(lgts, bnacc, Nn);
    bn_final_kernel<<<wblk, 256, 0, stream>>>(lgts, bnacc, bn_g, bn_b, (float*)d_out, Nn);
}
